// MultiHeadSelfAttention_16879221473629
// MI455X (gfx1250) — matrix operand order repaired and checked by reading, not compiled, not run on hardware
//
#include <hip/hip_runtime.h>

typedef __attribute__((ext_vector_type(16))) _Float16 v16h;
typedef __attribute__((ext_vector_type(8)))  _Float16 v8h;
typedef __attribute__((ext_vector_type(8)))  float    v8f;

#define BATCH 2
#define SEQ   2048
#define EMB   1024
#define NH    16
#define HD    64
#define MROWS (BATCH*SEQ)   // 4096

__device__ __forceinline__ v16h combine8(v8h lo, v8h hi) {
  return __builtin_shufflevector(lo, hi, 0,1,2,3,4,5,6,7,8,9,10,11,12,13,14,15);
}

// ---------------- conversion kernels ----------------
__global__ void cvt_f32_to_f16(const float* __restrict__ in, _Float16* __restrict__ out, int n) {
  int i = blockIdx.x * blockDim.x + threadIdx.x;
  if (i < n) out[i] = (_Float16)in[i];
}

// W [K,N] f32 row-major  ->  WT [N,K] f16 row-major
__global__ void transpose_cvt(const float* __restrict__ W, _Float16* __restrict__ WT, int K, int N) {
  int i = blockIdx.x * blockDim.x + threadIdx.x;
  if (i < K * N) {
    int k = i / N, n = i - k * N;
    WT[(size_t)n * K + k] = (_Float16)W[i];
  }
}

// ---------------- WMMA GEMM: C[M,N] = A[M,K] * B[K,N], B given as BT[N,K] ----------------
// Wave tile: 32 (M) x 64 (N). Each B fragment feeds two M-subtiles. Block = 4 waves.
template <bool F16OUT>
__global__ void __launch_bounds__(128) gemm_wmma(const _Float16* __restrict__ A,
                                                 const _Float16* __restrict__ BT,
                                                 void* __restrict__ Cout,
                                                 int M, int N, int K) {
  const int lane = threadIdx.x & 31;
  const int l16  = lane & 15;
  const int half = lane >> 4;
  const int wid  = blockIdx.x * (blockDim.x >> 5) + (threadIdx.x >> 5);
  const int ntiles = N >> 6;
  const int mt = wid / ntiles;          // 32-row tiles
  const int nt = wid - mt * ntiles;     // 64-col tiles
  if (mt * 32 >= M) return;

  v8f acc[2][4];
#pragma unroll
  for (int m = 0; m < 2; ++m)
#pragma unroll
    for (int f = 0; f < 4; ++f) acc[m][f] = (v8f){};

  const _Float16* arow0 = A + (size_t)(mt * 32 + l16) * K + half * 16;
  const _Float16* arow1 = arow0 + (size_t)16 * K;
  const _Float16* bcol  = BT + (size_t)(nt * 64 + l16) * K + half * 16;

  for (int kc = 0; kc < K; kc += 32) {
    // A frags 16x32: element j -> k = kc + 16*half + j
    v16h af0 = combine8(*(const v8h*)(arow0 + kc), *(const v8h*)(arow0 + kc + 8));
    v16h af1 = combine8(*(const v8h*)(arow1 + kc), *(const v8h*)(arow1 + kc + 8));
#pragma unroll
    for (int f = 0; f < 4; ++f) {
      // B frag 32x16: element j -> k = kc + 16*half + j (contiguous rows of BT)
      v16h bf = *(const v16h*)(bcol + kc + (size_t)(16 * f) * K);
      acc[0][f] = __builtin_amdgcn_wmma_f32_16x16x32_f16(false, af0, false, bf, (short)0, acc[0][f], false, false);
      acc[1][f] = __builtin_amdgcn_wmma_f32_16x16x32_f16(false, af1, false, bf, (short)0, acc[1][f], false, false);
    }
  }

  // C layout: element r -> row = r + 8*half, col = l16
#pragma unroll
  for (int m = 0; m < 2; ++m) {
#pragma unroll
    for (int r = 0; r < 8; ++r) {
      size_t row = (size_t)(mt * 32 + m * 16 + r + 8 * half);
      size_t col = (size_t)(nt * 64 + l16);
      if (F16OUT) {
        _Float16* C = (_Float16*)Cout;
#pragma unroll
        for (int f = 0; f < 4; ++f) C[row * N + col + 16 * f] = (_Float16)acc[m][f][r];
      } else {
        float* C = (float*)Cout;
#pragma unroll
        for (int f = 0; f < 4; ++f) C[row * N + col + 16 * f] = acc[m][f][r];
      }
    }
  }
}

// ---------------- RoPE + repack: qkvh[M,3072] -> qp/kp [B,H,S,D], vT [B,H,D,S] ----------------
__global__ void rope_pack(const _Float16* __restrict__ qkvh,
                          _Float16* __restrict__ qp, _Float16* __restrict__ kp,
                          _Float16* __restrict__ vT) {
  int t = blockIdx.x * blockDim.x + threadIdx.x;   // BATCH*NH*SEQ*32 threads
  int i  = t & 31;               // rotary pair index 0..31
  int s  = (t >> 5) & (SEQ - 1);
  int bh = t >> 16;              // 32*2048 = 2^16
  if (bh >= BATCH * NH) return;
  int b = bh >> 4, h = bh & 15;

  size_t src = (size_t)(b * SEQ + s) * (3 * EMB) + h * 64 + 2 * i;
  float q1 = (float)qkvh[src],        q2 = (float)qkvh[src + 1];
  float k1 = (float)qkvh[src + 1024], k2 = (float)qkvh[src + 1025];
  float v1 = (float)qkvh[src + 2048], v2 = (float)qkvh[src + 2049];

  // inv_freq = 10000^(-(2i)/64)
  float inv = __expf(-(float)(2 * i) * (9.210340371976184f / 64.0f));
  float ang = (float)s * inv;
  float c = __cosf(ang), sn = __sinf(ang);

  size_t o = (size_t)(bh * SEQ + s) * HD + 2 * i;
  qp[o]     = (_Float16)(q1 * c - q2 * sn);
  qp[o + 1] = (_Float16)(q2 * c + q1 * sn);
  kp[o]     = (_Float16)(k1 * c - k2 * sn);
  kp[o + 1] = (_Float16)(k2 * c + k1 * sn);

  size_t vo = ((size_t)bh * HD + 2 * i) * SEQ + s;
  vT[vo]       = (_Float16)v1;
  vT[vo + SEQ] = (_Float16)v2;
}

// ---------------- Flash attention: one wave per (b,h, 16-query tile), 64-key chunks ----------------
__global__ void __launch_bounds__(128) flash_attn(const _Float16* __restrict__ qp,
                                                  const _Float16* __restrict__ kp,
                                                  const _Float16* __restrict__ vT,
                                                  _Float16* __restrict__ aout) {
  __shared__ __align__(32) _Float16 Pbuf[4][16 * 64];   // per-wave P tile (C-layout -> A-layout bounce)

  const int lane = threadIdx.x & 31;
  const int l16  = lane & 15;
  const int half = lane >> 4;
  const int slot = threadIdx.x >> 5;
  const int wid  = blockIdx.x * 4 + slot;
  const int qt = wid & 127;        // SEQ/16 = 128 query tiles
  const int bh = wid >> 7;         // 0..31
  const int b = bh >> 4, h = bh & 15;

  // Q A-frags over head-dim (contraction 64 = two chunks of 32)
  const _Float16* qrow = qp + ((size_t)bh * SEQ + qt * 16 + l16) * HD + half * 16;
  v16h qa0 = combine8(*(const v8h*)(qrow),      *(const v8h*)(qrow + 8));
  v16h qa1 = combine8(*(const v8h*)(qrow + 32), *(const v8h*)(qrow + 40));

  float mrow[8], lrow[8];
#pragma unroll
  for (int r = 0; r < 8; ++r) { mrow[r] = -3.0e38f; lrow[r] = 0.0f; }
  v8f o[4];
#pragma unroll
  for (int f = 0; f < 4; ++f) o[f] = (v8f){};
  const float scale = 0.125f;   // 1/sqrt(64)

  for (int kc = 0; kc < SEQ; kc += 64) {
    // scores S = Q(16x64) @ K^T(64x64): 4 column frags x 2 contraction chunks
    v8f sf[4];
    const _Float16* krow = kp + ((size_t)bh * SEQ + kc + l16) * HD + half * 16;
#pragma unroll
    for (int f = 0; f < 4; ++f) {
      const _Float16* kr = krow + (size_t)(16 * f) * HD;
      v16h kb0 = *(const v16h*)(kr);       // head-dim 0..31
      v16h kb1 = *(const v16h*)(kr + 32);  // head-dim 32..63
      v8f s = {};
      s = __builtin_amdgcn_wmma_f32_16x16x32_f16(false, qa0, false, kb0, (short)0, s, false, false);
      s = __builtin_amdgcn_wmma_f32_16x16x32_f16(false, qa1, false, kb1, (short)0, s, false, false);
      sf[f] = s;
    }

    // online softmax over 64 columns: rows r+8*half live across the 16 lanes of this half
    float cr[8], pv[4][8];
#pragma unroll
    for (int r = 0; r < 8; ++r) {
      float a0 = sf[0][r] * scale, a1 = sf[1][r] * scale;
      float a2 = sf[2][r] * scale, a3 = sf[3][r] * scale;
      float cm = fmaxf(fmaxf(a0, a1), fmaxf(a2, a3));
#pragma unroll
      for (int off = 1; off < 16; off <<= 1) cm = fmaxf(cm, __shfl_xor(cm, off, 32));
      float mn = fmaxf(mrow[r], cm);
      float corr = __expf(mrow[r] - mn);
      float p0 = __expf(a0 - mn), p1 = __expf(a1 - mn);
      float p2 = __expf(a2 - mn), p3 = __expf(a3 - mn);
      float ps = (p0 + p1) + (p2 + p3);
#pragma unroll
      for (int off = 1; off < 16; off <<= 1) ps += __shfl_xor(ps, off, 32);
      lrow[r] = lrow[r] * corr + ps;
      mrow[r] = mn;
      cr[r] = corr;
      pv[0][r] = p0; pv[1][r] = p1; pv[2][r] = p2; pv[3][r] = p3;
    }

    // C-layout -> A-layout transpose of P (16x64) through LDS
#pragma unroll
    for (int r = 0; r < 8; ++r) {
#pragma unroll
      for (int f = 0; f < 4; ++f)
        Pbuf[slot][(r + 8 * half) * 64 + 16 * f + l16] = (_Float16)pv[f][r];
    }
    __syncthreads();
    const _Float16* pr = &Pbuf[slot][l16 * 64 + half * 16];
    v16h pa0 = combine8(*(const v8h*)(pr),      *(const v8h*)(pr + 8));    // keys 0..31
    v16h pa1 = combine8(*(const v8h*)(pr + 32), *(const v8h*)(pr + 40));   // keys 32..63
    __syncthreads();

    // rescale O, then O += P(16x64) @ V(64x64); V B-frags contiguous keys from vT
#pragma unroll
    for (int f = 0; f < 4; ++f)
#pragma unroll
      for (int r = 0; r < 8; ++r) o[f][r] *= cr[r];

    const _Float16* vrow = vT + ((size_t)bh * HD + l16) * SEQ + kc + half * 16;
#pragma unroll
    for (int f = 0; f < 4; ++f) {
      const _Float16* vr = vrow + (size_t)(16 * f) * SEQ;
      v16h vb0 = *(const v16h*)(vr);       // keys chunk 0
      v16h vb1 = *(const v16h*)(vr + 32);  // keys chunk 1
      o[f] = __builtin_amdgcn_wmma_f32_16x16x32_f16(false, pa0, false, vb0, (short)0, o[f], false, false);
      o[f] = __builtin_amdgcn_wmma_f32_16x16x32_f16(false, pa1, false, vb1, (short)0, o[f], false, false);
    }
  }

  // normalize and store packed [B,S,H*D] f16 for the out-projection GEMM
#pragma unroll
  for (int r = 0; r < 8; ++r) {
    float inv = 1.0f / lrow[r];
    int srow = qt * 16 + r + 8 * half;
    _Float16* op = aout + ((size_t)(b * SEQ + srow) * NH + h) * HD + l16;
#pragma unroll
    for (int f = 0; f < 4; ++f) op[16 * f] = (_Float16)(o[f][r] * inv);
  }
}

// ---------------- host launch ----------------
extern "C" void kernel_launch(void* const* d_in, const int* in_sizes, int n_in,
                              void* d_out, int out_size, void* d_ws, size_t ws_size,
                              hipStream_t stream) {
  (void)in_sizes; (void)n_in; (void)out_size; (void)ws_size;
  const float* x    = (const float*)d_in[0];   // [2,2048,1024]
  const float* Wqkv = (const float*)d_in[1];   // [1024,3072]
  const float* Wout = (const float*)d_in[2];   // [1024,1024]

  char* ws = (char*)d_ws;
  _Float16* xh    = (_Float16*)(ws + 0);          //  8 MiB: x in f16 [4096,1024]
  _Float16* wqkvT = (_Float16*)(ws + 8388608);    //  6 MiB: Wqkv^T f16 [3072,1024]
  _Float16* woutT = (_Float16*)(ws + 14680064);   //  2 MiB: Wout^T f16 [1024,1024]
  _Float16* qkvh  = (_Float16*)(ws + 16777216);   // 24 MiB: raw qkv f16 [4096,3072]
  _Float16* aout  = (_Float16*)(ws + 16777216);   //  8 MiB: attn out f16 (reuses dead qkvh)
  _Float16* qp    = (_Float16*)(ws + 41943040);   //  8 MiB: q packed [B,H,S,D]
  _Float16* kp    = (_Float16*)(ws + 50331648);   //  8 MiB: k packed [B,H,S,D]
  _Float16* vT    = (_Float16*)(ws + 58720256);   //  8 MiB: v transposed [B,H,D,S]
  // total: 64 MiB

  cvt_f32_to_f16<<<(MROWS * EMB + 255) / 256, 256, 0, stream>>>(x, xh, MROWS * EMB);
  transpose_cvt<<<(EMB * 3 * EMB + 255) / 256, 256, 0, stream>>>(Wqkv, wqkvT, EMB, 3 * EMB);
  transpose_cvt<<<(EMB * EMB + 255) / 256, 256, 0, stream>>>(Wout, woutT, EMB, EMB);

  // QKV projection: [4096,1024] x [1024,3072] -> f16 [4096,3072]; 32x64 wave tiles
  gemm_wmma<true><<<(MROWS / 32) * (3 * EMB / 64) / 4, 128, 0, stream>>>(
      xh, wqkvT, (void*)qkvh, MROWS, 3 * EMB, EMB);

  rope_pack<<<(BATCH * NH * SEQ * 32) / 256, 256, 0, stream>>>(qkvh, qp, kp, vT);

  // 4096 waves: one per (b,h,q-tile)
  flash_attn<<<(BATCH * NH * (SEQ / 16)) / 4, 128, 0, stream>>>(qp, kp, vT, aout);

  // out projection: [4096,1024] x [1024,1024] -> f32 d_out; 32x64 wave tiles
  gemm_wmma<false><<<(MROWS / 32) * (EMB / 64) / 4, 128, 0, stream>>>(
      aout, woutT, d_out, MROWS, EMB, EMB);
}
